// StochasticDurationPredictor_41180146434303
// MI455X (gfx1250) — matrix operand order repaired and checked by reading, not compiled, not run on hardware
//
#include <hip/hip_runtime.h>
#include <math.h>

// ---------------------------------------------------------------------------
// CDNA5 (gfx1250) wave32 WMMA types
// ---------------------------------------------------------------------------
typedef __attribute__((ext_vector_type(16))) __bf16 bf16x16;
typedef __attribute__((ext_vector_type(8)))  __bf16 bf16x8;
typedef __attribute__((ext_vector_type(4)))  __bf16 bf16x4;
typedef __attribute__((ext_vector_type(8)))  float  f32x8;
typedef __attribute__((ext_vector_type(4)))  float  f32x4;

#define GEMM_TILE 64
#define GEMM_KB   32
#define GEMM_PAD  8    // LDS row pad (elements); stride 40*2B=80B, 16B aligned

// ---------------------------------------------------------------------------
// Tiled GEMM: Y[b,o,t] = sum_c W[o,c] * X[b,c,t] + bias[o]   (optional *mask)
// f32 global operands -> bf16 WMMA with f32 accumulate.
// Block: 128 threads = 4 waves (2x2), each wave owns a 32x32 output sub-tile
// (2x2 tiles of v_wmma_f32_16x16x32_bf16). Software-pipelined K loop:
// next tile's global loads (vector f32x4) overlap WMMA on the current tile.
// ---------------------------------------------------------------------------
__global__ __launch_bounds__(128)
void k_gemm_wmma(const float* __restrict__ W, const float* __restrict__ X,
                 const float* __restrict__ bias, const float* __restrict__ mask,
                 float* __restrict__ Y, int O, int Kd, int T, int maskmul)
{
    __shared__ __bf16 sA[GEMM_TILE][GEMM_KB + GEMM_PAD];   // [o][k]
    __shared__ __bf16 sB[GEMM_TILE][GEMM_KB + GEMM_PAD];   // [t][k]

    const int b    = blockIdx.z;
    const int n0   = blockIdx.x * GEMM_TILE;   // t base
    const int m0   = blockIdx.y * GEMM_TILE;   // o base
    const int tid  = threadIdx.x;
    const int lane = tid & 31;
    const int wave = tid >> 5;
    const int wr   = wave >> 1;                // wave row (0..1)
    const int wc   = wave & 1;                 // wave col (0..1)
    const int half = lane >> 4;                // 0 or 1
    const int l15  = lane & 15;

    // Staging decomposition (per thread, 16 elements each):
    // A: row rA = tid>>1, k-chunk kcA = (tid&1)*16  -> 4x f32x4 along K
    const int rA  = tid >> 1;
    const int kcA = (tid & 1) << 4;
    const int oA  = m0 + rA;
    const bool aLive = (oA < O);
    // B: k-quad kqB = (tid>>4)*4, t-quad tqB = (tid&15)*4 -> 4x f32x4 along T
    const int kqB = (tid >> 4) << 2;
    const int tqB = (tid & 15) << 2;

    const float* Xb = X + (size_t)b * (size_t)Kd * (size_t)T;

    const f32x8 vzero = {0.f,0.f,0.f,0.f,0.f,0.f,0.f,0.f};
    f32x8 acc[2][2];
    acc[0][0] = vzero; acc[0][1] = vzero; acc[1][0] = vzero; acc[1][1] = vzero;

    f32x4 regA[4], regB[4];
    const f32x4 z4 = {0.f, 0.f, 0.f, 0.f};

    // ---- load tile k0 into registers (vector global loads) ----
    auto loadTile = [&](int k0) {
        if (aLive) {
            const float* wp = W + (size_t)oA * Kd + (k0 + kcA);
#pragma unroll
            for (int j = 0; j < 4; ++j) regA[j] = *(const f32x4*)(wp + j * 4);
        } else {
#pragma unroll
            for (int j = 0; j < 4; ++j) regA[j] = z4;
        }
        const float* xp = Xb + (size_t)(k0 + kqB) * T + (n0 + tqB);
#pragma unroll
        for (int j = 0; j < 4; ++j) regB[j] = *(const f32x4*)(xp + (size_t)j * T);
        if (k0 + 2 * GEMM_KB < Kd)   // prefetch 2 tiles ahead -> global_prefetch_b8
            __builtin_prefetch(Xb + (size_t)(k0 + 2 * GEMM_KB + kqB) * T + n0 + tqB, 0, 1);
    };

    // ---- convert registers -> bf16 LDS tiles (vector ds stores) ----
    auto storeTile = [&]() {
        bf16x8 v0, v1;
#pragma unroll
        for (int e = 0; e < 4; ++e) {
            v0[e]     = (__bf16)regA[0][e];
            v0[e + 4] = (__bf16)regA[1][e];
            v1[e]     = (__bf16)regA[2][e];
            v1[e + 4] = (__bf16)regA[3][e];
        }
        *(bf16x8*)&sA[rA][kcA]     = v0;   // b128 store
        *(bf16x8*)&sA[rA][kcA + 8] = v1;   // b128 store
#pragma unroll
        for (int e = 0; e < 4; ++e) {      // register transpose: [k][t] -> [t][k]
            bf16x4 pk;
#pragma unroll
            for (int j = 0; j < 4; ++j) pk[j] = (__bf16)regB[j][e];
            *(bf16x4*)&sB[tqB + e][kqB] = pk;   // b64 store
        }
    };

    loadTile(0);
    storeTile();

    for (int k0 = 0; k0 < Kd; k0 += GEMM_KB) {
        const bool more = (k0 + GEMM_KB) < Kd;
        __syncthreads();                    // staged tile visible
        if (more) loadTile(k0 + GEMM_KB);   // overlap next loads with WMMA

        // Fragments per CDNA5 wave32 layouts.
        // A (16x32): lanes 0-15 row M=l15 hold K 0..7 (v0-3) & 16..23 (v4-7);
        // lanes 16-31 hold K 8..15 & 24..31.
        bf16x16 afr[2], bfr[2];
#pragma unroll
        for (int i2 = 0; i2 < 2; ++i2) {
            const __bf16* p = &sA[wr * 32 + i2 * 16 + l15][half * 8];
            bf16x8 lo = *(const bf16x8*)(p);        // ds_load_b128
            bf16x8 hi = *(const bf16x8*)(p + 16);   // ds_load_b128
            bf16x16 av;
#pragma unroll
            for (int e = 0; e < 8; ++e) { av[e] = lo[e]; av[e + 8] = hi[e]; }
            afr[i2] = av;
        }
        // B (32x16): lane column N=l15; lanes 0-15 hold K 0..7 & 16..23,
        // lanes 16-31 hold K 8..15 & 24..31 (same order as A).
#pragma unroll
        for (int j2 = 0; j2 < 2; ++j2) {
            const __bf16* p = &sB[wc * 32 + j2 * 16 + l15][half * 8];
            bf16x8 lo = *(const bf16x8*)(p);
            bf16x8 hi = *(const bf16x8*)(p + 16);
            bf16x16 bv;
#pragma unroll
            for (int e = 0; e < 8; ++e) { bv[e] = lo[e]; bv[e + 8] = hi[e]; }
            bfr[j2] = bv;
        }
#pragma unroll
        for (int i2 = 0; i2 < 2; ++i2)
#pragma unroll
            for (int j2 = 0; j2 < 2; ++j2)
                acc[i2][j2] = __builtin_amdgcn_wmma_f32_16x16x32_bf16(
                    false, afr[i2], false, bfr[j2],
                    (short)0, acc[i2][j2], false, false);
        __syncthreads();                    // done reading LDS
        if (more) storeTile();              // stage the prefetched tile
    }

    // Epilogue: bias add (+ optional mask), store per 16x16 C/D layout:
    // VGPR r of lane holds (M = r + 8*half, N = l15).
    float* Yb = Y + (size_t)b * (size_t)O * (size_t)T;
#pragma unroll
    for (int i2 = 0; i2 < 2; ++i2) {
#pragma unroll
        for (int j2 = 0; j2 < 2; ++j2) {
            int tcol = n0 + wc * 32 + j2 * 16 + l15;
            float mk = maskmul ? mask[(size_t)b * T + tcol] : 1.f;
#pragma unroll
            for (int r = 0; r < 8; ++r) {
                int orow = m0 + wr * 32 + i2 * 16 + half * 8 + r;
                if (orow < O) {
                    float v = acc[i2][j2][r] + bias[orow];
                    Yb[(size_t)orow * T + tcol] = v * mk;
                }
            }
        }
    }
}

// ---------------------------------------------------------------------------
// Fused: masked depthwise conv (K=3, dilation dil) + bias -> LayerNorm(C)
// -> exact GELU.  One 256-thread block per (b,t) column; C = 768 = 3*256.
// ---------------------------------------------------------------------------
__global__ __launch_bounds__(256)
void k_dw_ln_gelu(const float* __restrict__ X, const float* __restrict__ mask,
                  const float* __restrict__ sw, const float* __restrict__ sb,
                  const float* __restrict__ gamma, const float* __restrict__ beta,
                  float* __restrict__ Y, int C, int T, int dil)
{
    const int t = blockIdx.x, b = blockIdx.y, tid = threadIdx.x;
    const float* Xb = X + (size_t)b * C * T;
    const float* mb = mask + (size_t)b * T;
    const int tm = t - dil, tp = t + dil;
    const float mC = mb[t];
    const float mL = (tm >= 0) ? mb[tm] : 0.f;
    const float mR = (tp < T) ? mb[tp] : 0.f;

    float vloc[3];
    float s = 0.f, sq = 0.f;
#pragma unroll
    for (int r = 0; r < 3; ++r) {
        int c = tid + (r << 8);
        const float* row = Xb + (size_t)c * T;
        float xl = (tm >= 0) ? row[tm] * mL : 0.f;
        float xc = row[t] * mC;
        float xr = (tp < T) ? row[tp] * mR : 0.f;
        float v = sw[c * 3 + 0] * xl + sw[c * 3 + 1] * xc + sw[c * 3 + 2] * xr + sb[c];
        vloc[r] = v; s += v; sq += v * v;
    }
    __shared__ float rs[256], rq[256];
    rs[tid] = s; rq[tid] = sq; __syncthreads();
    for (int off = 128; off > 0; off >>= 1) {
        if (tid < off) { rs[tid] += rs[tid + off]; rq[tid] += rq[tid + off]; }
        __syncthreads();
    }
    const float invC = 1.f / (float)C;
    float mean = rs[0] * invC;
    float var  = rq[0] * invC - mean * mean;
    float rstd = rsqrtf(var + 1e-5f);

    float* Yb = Y + (size_t)b * C * T;
#pragma unroll
    for (int r = 0; r < 3; ++r) {
        int c = tid + (r << 8);
        float u = (vloc[r] - mean) * rstd * gamma[c] + beta[c];
        u = 0.5f * u * (1.f + erff(u * 0.7071067811865475f));  // exact GELU
        Yb[(size_t)c * T + t] = u;
    }
}

// ---------------------------------------------------------------------------
// Fused: LayerNorm(C) of GEMM output -> GELU -> residual add (in place),
// optional end-of-DDS mask multiply.
// ---------------------------------------------------------------------------
__global__ __launch_bounds__(256)
void k_ln_gelu_add(const float* __restrict__ Yg, float* __restrict__ Xio,
                   const float* __restrict__ gamma, const float* __restrict__ beta,
                   const float* __restrict__ mask, int C, int T, int finalmask)
{
    const int t = blockIdx.x, b = blockIdx.y, tid = threadIdx.x;
    const float* Yb = Yg + (size_t)b * C * T;

    float vloc[3];
    float s = 0.f, sq = 0.f;
#pragma unroll
    for (int r = 0; r < 3; ++r) {
        int c = tid + (r << 8);
        float v = Yb[(size_t)c * T + t];
        vloc[r] = v; s += v; sq += v * v;
    }
    __shared__ float rs[256], rq[256];
    rs[tid] = s; rq[tid] = sq; __syncthreads();
    for (int off = 128; off > 0; off >>= 1) {
        if (tid < off) { rs[tid] += rs[tid + off]; rq[tid] += rq[tid + off]; }
        __syncthreads();
    }
    const float invC = 1.f / (float)C;
    float mean = rs[0] * invC;
    float var  = rq[0] * invC - mean * mean;
    float rstd = rsqrtf(var + 1e-5f);
    float mk = finalmask ? mask[(size_t)b * T + t] : 1.f;

    float* Xb = Xio + (size_t)b * C * T;
#pragma unroll
    for (int r = 0; r < 3; ++r) {
        int c = tid + (r << 8);
        float u = (vloc[r] - mean) * rstd * gamma[c] + beta[c];
        u = 0.5f * u * (1.f + erff(u * 0.7071067811865475f));
        float o = Xb[(size_t)c * T + t] + u;
        Xb[(size_t)c * T + t] = o * mk;
    }
}

// ---------------------------------------------------------------------------
// Flow pre-projection: X[b,c,t] = pre_w[c]*z[b,c0,t] + pre_b[c] + g[b,c,t]
// ---------------------------------------------------------------------------
__global__ __launch_bounds__(256)
void k_flow_pre(const float* __restrict__ Z, int c0,
                const float* __restrict__ pw, const float* __restrict__ pb,
                const float* __restrict__ G, float* __restrict__ X,
                int C, int T, int N)
{
    int i = blockIdx.x * 256 + threadIdx.x;
    if (i >= N) return;
    int t = i % T;
    int c = (i / T) % C;
    int b = i / (T * C);
    X[i] = pw[c] * Z[((size_t)b * 2 + c0) * T + t] + pb[c] + G[i];
}

__global__ __launch_bounds__(256)
void k_scale_z(const float* __restrict__ zin, const float* __restrict__ ns,
               float* __restrict__ Z, int N)
{
    int i = blockIdx.x * 256 + threadIdx.x;
    if (i < N) Z[i] = zin[i] * ns[0];
}

// ---------------------------------------------------------------------------
// Rational-quadratic spline inverse, one thread per (b,t).
// H is the masked flow projection, layout (B, 29, T).
// Writes x0*mask into channel c0 and spline(x1)*mask into channel c1.
// ---------------------------------------------------------------------------
__device__ __forceinline__ float softplusf(float x) {
    return (x > 20.f) ? x : log1pf(expf(x));
}

__device__ __forceinline__ void knots10(const float* un, float* cum, float* wid) {
    float mx = un[0];
#pragma unroll
    for (int k = 1; k < 10; ++k) mx = fmaxf(mx, un[k]);
    float e[10], ssum = 0.f;
#pragma unroll
    for (int k = 0; k < 10; ++k) { e[k] = expf(un[k] - mx); ssum += e[k]; }
    float inv = 1.f / ssum;
    float c = 0.f;
    cum[0] = -5.f;                         // padded 0 -> 2*TB*0 - TB = -TB
#pragma unroll
    for (int k = 0; k < 10; ++k) {
        float w = 0.001f + (1.f - 0.001f * 10.f) * e[k] * inv;
        c += w;
        cum[k + 1] = 10.f * c - 5.f;       // 2*TB*cumsum - TB
    }
    cum[10] = 5.f;                         // force last knot to +TB
#pragma unroll
    for (int k = 0; k < 10; ++k) wid[k] = cum[k + 1] - cum[k];
}

__global__ __launch_bounds__(256)
void k_spline(const float* __restrict__ H, float* __restrict__ Z,
              const float* __restrict__ mask, int T, int c0, int c1, int N)
{
    int i = blockIdx.x * 256 + threadIdx.x;
    if (i >= N) return;
    int b = i / T, t = i % T;
    const float scale = 0.03608439182435161f;  // 1/sqrt(768)

    const float* Hb = H + (size_t)b * 29 * T + t;
    float uw[10], uh[10], ud9[9];
#pragma unroll
    for (int k = 0; k < 10; ++k) uw[k] = Hb[(size_t)k * T] * scale;
#pragma unroll
    for (int k = 0; k < 10; ++k) uh[k] = Hb[(size_t)(10 + k) * T] * scale;
#pragma unroll
    for (int k = 0; k < 9; ++k)  ud9[k] = Hb[(size_t)(20 + k) * T];

    float x   = Z[((size_t)b * 2 + c1) * T + t];
    float x0v = Z[((size_t)b * 2 + c0) * T + t];

    float cw[11], wd[10], ch[11], hd[10], dv[11];
    knots10(uw, cw, wd);
    knots10(uh, ch, hd);
    dv[0] = 1.f; dv[10] = 1.f;  // md + softplus(log(exp(1-md)-1)) == 1 exactly
#pragma unroll
    for (int k = 0; k < 9; ++k) dv[k + 1] = 0.001f + softplusf(ud9[k]);

    bool inside = (x >= -5.f) && (x <= 5.f);
    float xs = fminf(fmaxf(x, -5.f), 5.f);

    int cnt = 0;
#pragma unroll
    for (int j = 0; j < 11; ++j) {
        float blv = (j == 10) ? (ch[10] + 1e-6f) : ch[j];
        cnt += (xs >= blv) ? 1 : 0;
    }
    int idx = cnt - 1;
    idx = (idx < 0) ? 0 : ((idx > 9) ? 9 : idx);

    float icw = cw[idx], ibw = wd[idx], ich = ch[idx], ih = hd[idx];
    float idel = ih / ibw;
    float dk = dv[idx], dk1 = dv[idx + 1];
    float dy = xs - ich;
    float sv = dk + dk1 - 2.f * idel;
    float av = dy * sv + ih * (idel - dk);
    float bv = ih * dk - dy * sv;
    float cv = -idel * dy;
    float disc = fmaxf(bv * bv - 4.f * av * cv, 0.f);
    float root = 2.f * cv / (-bv - sqrtf(disc));
    float outx = root * ibw + icw;
    float x1n = inside ? outx : x;

    float mk = mask[(size_t)b * T + t];
    Z[((size_t)b * 2 + c1) * T + t] = x1n * mk;
    Z[((size_t)b * 2 + c0) * T + t] = x0v * mk;
}

__global__ __launch_bounds__(256)
void k_final(const float* __restrict__ Z, const float* __restrict__ ea_m,
             const float* __restrict__ ea_logs, const float* __restrict__ mask,
             float* __restrict__ out, int T, int N)
{
    int i = blockIdx.x * 256 + threadIdx.x;
    if (i >= N) return;
    int b = i / T, t = i % T;
    float v = (Z[((size_t)b * 2) * T + t] - ea_m[0]) * expf(-ea_logs[0]);
    out[i] = v * mask[i];
}

// ---------------------------------------------------------------------------
// Host orchestration
// ---------------------------------------------------------------------------
extern "C" void kernel_launch(void* const* d_in, const int* in_sizes, int n_in,
                              void* d_out, int out_size, void* d_ws, size_t ws_size,
                              hipStream_t stream)
{
    (void)in_sizes; (void)n_in; (void)out_size; (void)ws_size;
    const int Bn = 32, C = 768, T = 768;
    const size_t nBCT = (size_t)Bn * C * T;

    const float* x       = (const float*)d_in[0];
    const float* mask    = (const float*)d_in[1];
    const float* zin     = (const float*)d_in[2];
    const float* ns      = (const float*)d_in[3];
    const float* pre_w   = (const float*)d_in[4];
    const float* pre_b   = (const float*)d_in[5];
    const float* proj_w  = (const float*)d_in[6];
    const float* proj_b  = (const float*)d_in[7];
    const float* ea_m    = (const float*)d_in[32];
    const float* ea_logs = (const float*)d_in[33];

    // Workspace carve (f32): 4 activation buffers + flow projection + z.
    float* Xb  = (float*)d_ws;
    float* Y1  = Xb + nBCT;
    float* Y2  = Y1 + nBCT;
    float* G   = Y2 + nBCT;
    float* P29 = G + nBCT;
    float* Z   = P29 + (size_t)Bn * 29 * T;

    dim3 gemmGrid(T / GEMM_TILE, C / GEMM_TILE, Bn);
    dim3 gemm29Grid(T / GEMM_TILE, 1, Bn);
    dim3 colGrid(T, Bn);

    // ---- main path: h = pre(x); h = dds(h); g = proj(h)*mask ----
    k_gemm_wmma<<<gemmGrid, 128, 0, stream>>>(pre_w, x, pre_b, mask, Xb, C, C, T, 0);
    for (int i = 0; i < 3; ++i) {
        int L = 8 + i * 8;
        int dil = 1; for (int q = 0; q < i; ++q) dil *= 3;
        k_dw_ln_gelu<<<colGrid, 256, 0, stream>>>(Xb, mask,
            (const float*)d_in[L], (const float*)d_in[L + 1],
            (const float*)d_in[L + 2], (const float*)d_in[L + 3], Y1, C, T, dil);
        k_gemm_wmma<<<gemmGrid, 128, 0, stream>>>((const float*)d_in[L + 4], Y1,
            (const float*)d_in[L + 5], mask, Y2, C, C, T, 0);
        k_ln_gelu_add<<<colGrid, 256, 0, stream>>>(Y2, Xb,
            (const float*)d_in[L + 6], (const float*)d_in[L + 7], mask, C, T,
            (i == 2) ? 1 : 0);
    }
    k_gemm_wmma<<<gemmGrid, 128, 0, stream>>>(proj_w, Xb, proj_b, mask, G, C, C, T, 1);

    // ---- flows (reverse): flows[2], flip, flows[1], flip, flows[0] ----
    int nz = Bn * 2 * T;
    k_scale_z<<<(nz + 255) / 256, 256, 0, stream>>>(zin, ns, Z, nz);

    int c0 = 0, c1 = 1;
    for (int f = 2; f >= 0; --f) {
        int base = 34 + f * 28;
        k_flow_pre<<<(int)((nBCT + 255) / 256), 256, 0, stream>>>(Z, c0,
            (const float*)d_in[base], (const float*)d_in[base + 1], G, Xb,
            C, T, (int)nBCT);
        for (int i = 0; i < 3; ++i) {
            int L = base + 2 + i * 8;
            int dil = 1; for (int q = 0; q < i; ++q) dil *= 3;
            k_dw_ln_gelu<<<colGrid, 256, 0, stream>>>(Xb, mask,
                (const float*)d_in[L], (const float*)d_in[L + 1],
                (const float*)d_in[L + 2], (const float*)d_in[L + 3], Y1, C, T, dil);
            k_gemm_wmma<<<gemmGrid, 128, 0, stream>>>((const float*)d_in[L + 4], Y1,
                (const float*)d_in[L + 5], mask, Y2, C, C, T, 0);
            k_ln_gelu_add<<<colGrid, 256, 0, stream>>>(Y2, Xb,
                (const float*)d_in[L + 6], (const float*)d_in[L + 7], mask, C, T,
                (i == 2) ? 1 : 0);
        }
        // skinny projection (O=29, zero-padded rows with guards)
        k_gemm_wmma<<<gemm29Grid, 128, 0, stream>>>((const float*)d_in[base + 26], Xb,
            (const float*)d_in[base + 27], mask, P29, 29, C, T, 1);
        int nbt = Bn * T;
        k_spline<<<(nbt + 255) / 256, 256, 0, stream>>>(P29, Z, mask, T, c0, c1, nbt);
        if (f != 0) { int tmp = c0; c0 = c1; c1 = tmp; }  // z = z[:, ::-1]
    }

    int nbt = Bn * T;
    k_final<<<(nbt + 255) / 256, 256, 0, stream>>>(Z, ea_m, ea_logs, mask,
                                                   (float*)d_out, T, nbt);
}
